// QuantLinearW4A8_17085379903908
// MI455X (gfx1250) — 8-bit matrix operand order repaired and checked by reading, compiled, not run on hardware
//
#include <hip/hip_runtime.h>
#include <hip/hip_bf16.h>

#define TOKENS      16384
#define IN_F        4096
#define OUT_F       4096
#define GS          128
#define GROUPS_ROW  (IN_F / GS)
#define ASTRIDE     144
#define BSTRIDE     144

typedef __attribute__((ext_vector_type(8))) int   v8i;
typedef __attribute__((ext_vector_type(8))) float v8f;

#if defined(__has_builtin)
#if __has_builtin(__builtin_amdgcn_global_load_async_to_lds_b128)
#define HAVE_ASYNC_LDS 1
#endif
#endif

#ifdef HAVE_ASYNC_LDS
typedef int v4i_vs __attribute__((vector_size(16)));
typedef __attribute__((address_space(1))) v4i_vs as1_v4i;
typedef __attribute__((address_space(3))) v4i_vs as3_v4i;
__device__ __forceinline__ void async_copy16(const void* g, void* l) {
    __builtin_amdgcn_global_load_async_to_lds_b128(
        (as1_v4i*)(uintptr_t)g,
        (as3_v4i*)(uintptr_t)(unsigned)(uintptr_t)l,
        0, 0);
}
__device__ __forceinline__ void async_wait0() {
#if __has_builtin(__builtin_amdgcn_s_wait_asynccnt)
    __builtin_amdgcn_s_wait_asynccnt(0);
#else
    asm volatile("s_wait_asynccnt 0" ::: "memory");
#endif
}
#else
__device__ __forceinline__ void async_copy16(const void* g, void* l) {
    *(int4*)l = *(const int4*)g;
}
__device__ __forceinline__ void async_wait0() {}
#endif

__global__ void k_init(unsigned int* amax) {
    if (threadIdx.x == 0 && blockIdx.x == 0) *amax = 0u;
}

__global__ __launch_bounds__(256) void k_amax(const float* __restrict__ x,
                                              unsigned int* __restrict__ amax,
                                              int n) {
    int gid    = blockIdx.x * blockDim.x + threadIdx.x;
    int stride = gridDim.x * blockDim.x * 4;
    float m = 0.0f;
    for (int idx = gid * 4; idx < n; idx += stride) {
        float4 v = *(const float4*)(x + idx);
        m = fmaxf(m, fmaxf(fmaxf(fabsf(v.x), fabsf(v.y)),
                           fmaxf(fabsf(v.z), fabsf(v.w))));
    }
    #pragma unroll
    for (int off = 16; off > 0; off >>= 1)
        m = fmaxf(m, __shfl_down(m, off, 32));
    __shared__ float smax[8];
    if ((threadIdx.x & 31) == 0) smax[threadIdx.x >> 5] = m;
    __syncthreads();
    if (threadIdx.x < 8) {
        m = smax[threadIdx.x];
        #pragma unroll
        for (int off = 4; off > 0; off >>= 1)
            m = fmaxf(m, __shfl_down(m, off, 8));
        if (threadIdx.x == 0) atomicMax(amax, __float_as_uint(m));
    }
}

__global__ __launch_bounds__(256) void k_quant(const float* __restrict__ x,
                                               const unsigned int* __restrict__ amax_bits,
                                               signed char* __restrict__ xq,
                                               float* __restrict__ rsf) {
    const int m   = blockIdx.x;
    const int tid = threadIdx.x;
    const float amax = __uint_as_float(*amax_bits);
    const float sx   = fmaxf(amax * (1.0f / 127.0f), 1e-8f);
    const float inv  = 1.0f / sx;

    const float* xr = x + (long long)m * IN_F + tid * 16;
    int* xo = (int*)(xq + (long long)m * IN_F + tid * 16);

    int lsum = 0;
    #pragma unroll
    for (int v4 = 0; v4 < 4; ++v4) {
        float4 v = *(const float4*)(xr + v4 * 4);
        int q0 = min(127, max(-128, __float2int_rn(v.x * inv)));
        int q1 = min(127, max(-128, __float2int_rn(v.y * inv)));
        int q2 = min(127, max(-128, __float2int_rn(v.z * inv)));
        int q3 = min(127, max(-128, __float2int_rn(v.w * inv)));
        lsum += q0 + q1 + q2 + q3;
        xo[v4] = (q0 & 255) | ((q1 & 255) << 8) | ((q2 & 255) << 16) | ((q3 & 255) << 24);
    }

    __shared__ int ssum[256];
    ssum[tid] = lsum;
    __syncthreads();
    if (tid < GROUPS_ROW) {
        int s = 0;
        #pragma unroll
        for (int i = 0; i < 8; ++i) s += ssum[tid * 8 + i];
        rsf[m * GROUPS_ROW + tid] = (float)s;
    }
}

__global__ __launch_bounds__(256) void k_gemm(const int* __restrict__ pw,
                                              const float* __restrict__ scales,
                                              const float* __restrict__ zeros,
                                              const signed char* __restrict__ xq,
                                              const float* __restrict__ rsf,
                                              const unsigned int* __restrict__ amax_bits,
                                              float* __restrict__ out) {
    __shared__ signed char lds_a[128 * ASTRIDE];
    __shared__ signed char lds_b[128 * BSTRIDE];

    const int tid   = threadIdx.x;
    const int lane  = tid & 31;
    const int wave  = tid >> 5;
    const int n0    = blockIdx.x * 128;
    const int m0    = blockIdx.y * 128;
    const int wm    = wave & 3;
    const int wn    = wave >> 2;
    const int mbase = m0 + wm * 32;
    const int nbase = n0 + wn * 64;
    const int nlane = lane & 15;
    const int hi    = (lane >> 4) & 1;

    const float amax = __uint_as_float(*amax_bits);
    const float sx   = fmaxf(amax * (1.0f / 127.0f), 1e-8f);

    v8f facc[2][4];
    #pragma unroll
    for (int mt = 0; mt < 2; ++mt)
        #pragma unroll
        for (int nt = 0; nt < 4; ++nt)
            #pragma unroll
            for (int r = 0; r < 8; ++r) facc[mt][nt][r] = 0.0f;

    const int arow  = tid >> 1;
    const int akoff = (tid & 1) * 64;

    for (int g = 0; g < GROUPS_ROW; ++g) {
        const int kbase = g * GS;

        {
            const signed char* gsrc = xq + (long long)(m0 + arow) * IN_F + kbase + akoff;
            signed char*       ldst = &lds_a[arow * ASTRIDE + akoff];
            #pragma unroll
            for (int c = 0; c < 4; ++c)
                async_copy16(gsrc + c * 16, ldst + c * 16);
        }

        for (int c = tid; c < 128 * 16; c += 256) {
            int nn = c >> 4;
            int kk = (c & 15) * 8;
            long long fidx = (long long)(n0 + nn) * IN_F + kbase + kk;
            long long pi   = fidx >> 1;
            int4 p = *(const int4*)(pw + pi);
            int w0 = ((p.x >> 4) & 15) | ((p.x & 15) << 8) |
                     (((p.y >> 4) & 15) << 16) | ((p.y & 15) << 24);
            int w1 = ((p.z >> 4) & 15) | ((p.z & 15) << 8) |
                     (((p.w >> 4) & 15) << 16) | ((p.w & 15) << 24);
            *(int2*)(&lds_b[nn * BSTRIDE + kk]) = make_int2(w0, w1);
            if (g + 1 < GROUPS_ROW)
                __builtin_prefetch(&pw[pi + (GS >> 1)], 0, 1);
        }

        async_wait0();
        __syncthreads();

        v8i afrag[2][2];
        #pragma unroll
        for (int mt = 0; mt < 2; ++mt) {
            const signed char* ar = &lds_a[(wm * 32 + mt * 16 + nlane) * ASTRIDE];
            #pragma unroll
            for (int h = 0; h < 2; ++h)
                #pragma unroll
                for (int c = 0; c < 4; ++c) {
                    int2 d = *(const int2*)(ar + h * 64 + (c >> 1) * 32 + hi * 16 + (c & 1) * 8);
                    afrag[mt][h][2 * c]     = d.x;
                    afrag[mt][h][2 * c + 1] = d.y;
                }
        }

        float rvf[2][8];
        #pragma unroll
        for (int mt = 0; mt < 2; ++mt)
            #pragma unroll
            for (int r = 0; r < 8; ++r)
                rvf[mt][r] = rsf[(mbase + mt * 16 + r + 8 * hi) * GROUPS_ROW + g];

        #pragma unroll
        for (int nt = 0; nt < 4; ++nt) {
            const signed char* bb = &lds_b[(wn * 64 + nt * 16 + nlane) * BSTRIDE];
            v8i bfrag[2];
            #pragma unroll
            for (int h = 0; h < 2; ++h) {
                int4 p0 = *(const int4*)(bb + h * 64 + hi * 16);
                int4 p1 = *(const int4*)(bb + h * 64 + 32 + hi * 16);
                bfrag[h][0] = p0.x; bfrag[h][1] = p0.y;
                bfrag[h][2] = p0.z; bfrag[h][3] = p0.w;
                bfrag[h][4] = p1.x; bfrag[h][5] = p1.y;
                bfrag[h][6] = p1.z; bfrag[h][7] = p1.w;
            }

            v8i zc = {0, 0, 0, 0, 0, 0, 0, 0};
            v8i i0 = __builtin_amdgcn_wmma_i32_16x16x64_iu8(true, afrag[0][0], false, bfrag[0],
                                                            zc, false, false);
            v8i i1 = __builtin_amdgcn_wmma_i32_16x16x64_iu8(true, afrag[1][0], false, bfrag[0],
                                                            zc, false, false);
            i0 = __builtin_amdgcn_wmma_i32_16x16x64_iu8(true, afrag[0][1], false, bfrag[1],
                                                        i0, false, false);
            i1 = __builtin_amdgcn_wmma_i32_16x16x64_iu8(true, afrag[1][1], false, bfrag[1],
                                                        i1, false, false);

            const int n = nbase + nt * 16 + nlane;
            const float s = scales[n * GROUPS_ROW + g];
            const float z = zeros[n * GROUPS_ROW + g];
            #pragma unroll
            for (int r = 0; r < 8; ++r)
                facc[0][nt][r] += s * ((float)i0[r] - z * rvf[0][r]);
            #pragma unroll
            for (int r = 0; r < 8; ++r)
                facc[1][nt][r] += s * ((float)i1[r] - z * rvf[1][r]);
        }
        __syncthreads();
    }

    #pragma unroll
    for (int mt = 0; mt < 2; ++mt)
        #pragma unroll
        for (int nt = 0; nt < 4; ++nt) {
            const int n = nbase + nt * 16 + nlane;
            #pragma unroll
            for (int r = 0; r < 8; ++r) {
                const int m = mbase + mt * 16 + r + 8 * hi;
                out[(long long)m * OUT_F + n] = sx * facc[mt][nt][r];
            }
        }
}

extern "C" void kernel_launch(void* const* d_in, const int* in_sizes, int n_in,
                              void* d_out, int out_size, void* d_ws, size_t ws_size,
                              hipStream_t stream) {
    const float* x      = (const float*)d_in[0];
    const int*   pw     = (const int*)d_in[1];
    const float* scales = (const float*)d_in[2];
    const float* zeros  = (const float*)d_in[3];
    float*       out    = (float*)d_out;

    unsigned int* d_amax = (unsigned int*)d_ws;
    float*        d_rsf  = (float*)((char*)d_ws + 256);
    signed char*  d_xq   = (signed char*)((char*)d_ws + 256 +
                                          (size_t)TOKENS * GROUPS_ROW * sizeof(float));

    k_init<<<1, 1, 0, stream>>>(d_amax);
    k_amax<<<1024, 256, 0, stream>>>(x, d_amax, TOKENS * IN_F);
    k_quant<<<TOKENS, 256, 0, stream>>>(x, d_amax, d_xq, d_rsf);

    dim3 grid(OUT_F / 128, TOKENS / 128);
    k_gemm<<<grid, 256, 0, stream>>>(pw, scales, zeros, d_xq, d_rsf, d_amax, out);
}
